// Mamba2Mixer_43525198577755
// MI455X (gfx1250) — matrix operand order repaired and checked by reading, not compiled, not run on hardware
//
#include <hip/hip_runtime.h>
#include <hip/hip_bf16.h>

typedef __bf16 bf16;
typedef __attribute__((ext_vector_type(16))) __bf16 bf16x16;
typedef __attribute__((ext_vector_type(8)))  __bf16 bf16x8;
typedef __attribute__((ext_vector_type(8)))  float  v8f;

#define HIDN   1024
#define INTER  2048
#define NH     32
#define HP     64
#define NS     128
#define CHK    256
#define CONVD  (INTER + 2*NS)         // 2304
#define PROJD  (INTER + CONVD + NH)   // 4384
#define PROJP  4480                   // PROJD rounded up to 128
#define BB     2
#define SEQ    4096
#define BS     (BB*SEQ)               // 8192
#define NC     (SEQ/CHK)              // 16

// ---------------- WMMA helpers (gfx1250, wave32) ----------------

__device__ inline v8f wmma_bf16(bf16x16 a, bf16x16 b, v8f c) {
  return __builtin_amdgcn_wmma_f32_16x16x32_bf16(false, a, false, b, (short)0, c, false, false);
}

// A fragment 16x32 (MxK) from row-major bf16 [ld elems/row].
// Lane L: M = L&15, half = L>>4. elems 0..7 -> K = half*8..+7, elems 8..15 -> K = 16+half*8..+7
__device__ inline bf16x16 frag_a(const bf16* base, int ld) {
  int lane = threadIdx.x & 31;
  int m = lane & 15, hf = lane >> 4;
  bf16x8 lo = *(const bf16x8*)(base + m*ld + hf*8);
  bf16x8 hi = *(const bf16x8*)(base + m*ld + 16 + hf*8);
  bf16x16 r;
#pragma unroll
  for (int i = 0; i < 8; ++i) { r[i] = lo[i]; r[8+i] = hi[i]; }
  return r;
}

// B fragment 32x16 (KxN) from K-contiguous layout [n][k] (row stride ld elems).
// Lane L: N = L&15, half = L>>4. elems 0..7 -> K = half*8..+7, elems 8..15 -> K = 16+half*8..+7
__device__ inline bf16x16 frag_bt(const bf16* base, int ld) {
  int lane = threadIdx.x & 31;
  int n = lane & 15, hf = lane >> 4;
  const bf16* p = base + n*ld;
  bf16x8 lo = *(const bf16x8*)(p + hf*8);
  bf16x8 hi = *(const bf16x8*)(p + 16 + hf*8);
  bf16x16 r;
#pragma unroll
  for (int i = 0; i < 8; ++i) { r[i] = lo[i]; r[8+i] = hi[i]; }
  return r;
}

// ---------------- bf16 converters ----------------

__global__ void k_cvt(const float* __restrict__ in, bf16* __restrict__ out, size_t n) {
  size_t i = (size_t)blockIdx.x * 256 + threadIdx.x;
  if (i < n) out[i] = (bf16)in[i];
}

// W [K,N] f32 (row-major) -> WT [NP,K] bf16, rows n>=N zero-filled
__global__ void k_cvtT(const float* __restrict__ in, bf16* __restrict__ out,
                       int K, int N, int NP) {
  size_t i = (size_t)blockIdx.x * 256 + threadIdx.x;
  if (i >= (size_t)NP * K) return;
  int k = (int)(i % K);
  int n = (int)(i / K);
  out[i] = (bf16)((n < N) ? in[(size_t)k * N + n] : 0.f);
}

// ---------------- Async double-buffered WMMA GEMM ----------------
// C[M,N] f32 = A[M,K] bf16 * BT[N,K]^T bf16   (BT pre-transposed, rows padded to >= gridDim.x*128)

__launch_bounds__(256)
__global__ void k_gemm_async(const bf16* __restrict__ A, const bf16* __restrict__ BT,
                             float* __restrict__ C, int M, int N, int K) {
  __shared__ __align__(16) bf16 As[2][128*32];  // [m][k]
  __shared__ __align__(16) bf16 Bs[2][128*32];  // [n][k]
  int t = threadIdx.x;
  int wid = t >> 5, wm = wid & 3, wn = wid >> 2;
  int m0 = blockIdx.y * 128, n0 = blockIdx.x * 128;

  v8f acc[2][4];
#pragma unroll
  for (int i = 0; i < 2; ++i)
#pragma unroll
    for (int j = 0; j < 4; ++j)
#pragma unroll
      for (int r = 0; r < 8; ++r) acc[i][j][r] = 0.f;

  // Issue one K-tile's worth of async global->LDS copies (4 async ops per wave).
  auto stage = [&](int buf, int k0) {
#pragma unroll
    for (int ii = 0; ii < 2; ++ii) {
      int i = t + ii * 256;
      int row = i >> 2, seg = i & 3;                 // 4 x 16B segments per 64B row
      uint32_t la = (uint32_t)(uintptr_t)&As[buf][row*32 + seg*8];
      uint64_t ga = (uint64_t)(uintptr_t)(A + (size_t)(m0 + row)*K + k0 + seg*8);
      asm volatile("global_load_async_to_lds_b128 %0, %1, off"
                   :: "v"(la), "v"(ga) : "memory");
      uint32_t lb = (uint32_t)(uintptr_t)&Bs[buf][row*32 + seg*8];
      uint64_t gb = (uint64_t)(uintptr_t)(BT + (size_t)(n0 + row)*K + k0 + seg*8);
      asm volatile("global_load_async_to_lds_b128 %0, %1, off"
                   :: "v"(lb), "v"(gb) : "memory");
    }
  };

  stage(0, 0);
  const int nk = K >> 5;
  for (int kt = 0; kt < nk; ++kt) {
    int buf = kt & 1;
    if (kt + 1 < nk) {
      stage(buf ^ 1, (kt + 1) * 32);
      asm volatile("s_wait_asynccnt 0x4" ::: "memory");  // current tile's 4 copies done
    } else {
      asm volatile("s_wait_asynccnt 0x0" ::: "memory");
    }
    __syncthreads();
    bf16x16 af[2], bfg[4];
#pragma unroll
    for (int i = 0; i < 2; ++i) af[i] = frag_a(&As[buf][(wm*32 + i*16)*32], 32);
#pragma unroll
    for (int j = 0; j < 4; ++j) bfg[j] = frag_bt(&Bs[buf][(wn*64 + j*16)*32], 32);
#pragma unroll
    for (int i = 0; i < 2; ++i)
#pragma unroll
      for (int j = 0; j < 4; ++j)
        acc[i][j] = wmma_bf16(af[i], bfg[j], acc[i][j]);
    __syncthreads();   // all waves done reading this buffer before it is refilled
  }

  int lane = t & 31, nc16 = lane & 15, hf = lane >> 4;
#pragma unroll
  for (int i = 0; i < 2; ++i)
#pragma unroll
    for (int j = 0; j < 4; ++j) {
      int col = n0 + wn*64 + j*16 + nc16;
      if (col < N) {
#pragma unroll
        for (int r = 0; r < 8; ++r)
          C[(size_t)(m0 + wm*32 + i*16 + r + 8*hf)*N + col] = acc[i][j][r];
      }
    }
}

// ---------------- Depthwise causal conv (K=4) + bias + SiLU ----------------

__global__ void k_conv(const float* __restrict__ proj, const float* __restrict__ cw,
                       const float* __restrict__ cb, float* __restrict__ out) {
  size_t idx = (size_t)blockIdx.x * 256 + threadIdx.x;
  if (idx >= (size_t)BS * CONVD) return;
  int ch = (int)(idx % CONVD);
  size_t bs = idx / CONVD;
  int s = (int)(bs % SEQ);
  size_t brow = bs - s;
  float acc = cb[ch];
#pragma unroll
  for (int j = 0; j < 4; ++j) {
    int st = s - 3 + j;
    if (st >= 0) acc += cw[ch*4 + j] * proj[(brow + st)*PROJD + INTER + ch];
  }
  out[idx] = acc / (1.f + __expf(-acc));   // silu
}

// ---------------- dt softplus + per-chunk cumsum of A*dt ----------------

__global__ void k_dtscan(const float* __restrict__ proj, const float* __restrict__ dt_bias,
                         const float* __restrict__ A_log, float* __restrict__ dtq,
                         float* __restrict__ acs_g, float* __restrict__ alast_g) {
  __shared__ float buf[256];
  int blk = blockIdx.x;
  int c = blk & 15, h = (blk >> 4) & 31, b = blk >> 9;
  int t = threadIdx.x;
  size_t bs = (size_t)b*SEQ + c*CHK + t;
  float x = proj[bs*PROJD + INTER + CONVD + h] + dt_bias[h];
  float sp = (x > 20.f) ? x : log1pf(__expf(x));
  sp = fminf(fmaxf(sp, 0.f), 100.f);
  dtq[bs*NH + h] = sp;
  float ad = -__expf(A_log[h]) * sp;
  buf[t] = ad;
  __syncthreads();
  for (int o = 1; o < 256; o <<= 1) {
    float v = (t >= o) ? buf[t - o] : 0.f;
    __syncthreads();
    buf[t] += v;
    __syncthreads();
  }
  acs_g[((size_t)(b*NH + h)*NC + c)*CHK + t] = buf[t];
  if (t == 255) alast_g[(size_t)(b*NH + h)*NC + c] = buf[255];
}

// ---------------- Per-(b,chunk,head): Y_diag + chunk states (WMMA) ----------------

__launch_bounds__(256)
__global__ void k_chunk(const float* __restrict__ conv_out, const float* __restrict__ dtq,
                        const float* __restrict__ acs_g, float* __restrict__ ydiag,
                        float* __restrict__ cstates) {
  extern __shared__ __align__(16) char smem[];
  float* acs = (float*)smem;                  // [256]
  bf16*  XsT = (bf16*)(smem + 1024);          // [64][256]  X^T scaled by dt
  bf16*  Bsm = XsT + 64*256;                  // [256][128]
  bf16*  Csm = Bsm + 256*128;                 // [256][128]
  bf16*  Ssb = Csm + 256*128;                 // 8 * [32][32] wave-private score tiles
  int blk = blockIdx.x;
  int h = blk & 31, c = (blk >> 5) & 15, b = blk >> 9;
  int t = threadIdx.x;
  size_t bs0 = (size_t)b*SEQ + c*CHK;

  acs[t] = acs_g[((size_t)(b*NH + h)*NC + c)*CHK + t];
  for (int i = t; i < 64*256; i += 256) {
    int p = i >> 8, s = i & 255;
    float xv = conv_out[(bs0 + s)*CONVD + h*HP + p];
    XsT[p*256 + s] = (bf16)(xv * dtq[(bs0 + s)*NH + h]);
  }
  for (int i = t; i < 256*128; i += 256) {
    int s = i >> 7, n = i & 127;
    Bsm[i] = (bf16)conv_out[(bs0 + s)*CONVD + INTER + n];
    Csm[i] = (bf16)conv_out[(bs0 + s)*CONVD + INTER + NS + n];
  }
  __syncthreads();

  int w = t >> 5, lane = t & 31, nc16 = lane & 15, hf = lane >> 4;
  int m0 = w * 32;
  bf16* Ss = Ssb + w*32*32;

  // Phase 1: Y_diag = (L o (C B^T)) X for this wave's 32 query rows
  v8f yacc[2][4];
#pragma unroll
  for (int i = 0; i < 2; ++i)
#pragma unroll
    for (int j = 0; j < 4; ++j)
#pragma unroll
      for (int r = 0; r < 8; ++r) yacc[i][j][r] = 0.f;

  for (int kt = 0; kt <= w; ++kt) {
    int s0 = kt * 32;
    v8f sacc[2][2];
#pragma unroll
    for (int i = 0; i < 2; ++i)
#pragma unroll
      for (int j = 0; j < 2; ++j)
#pragma unroll
        for (int r = 0; r < 8; ++r) sacc[i][j][r] = 0.f;
#pragma unroll
    for (int kk = 0; kk < 4; ++kk) {
      bf16x16 a0 = frag_a(&Csm[(m0     )*128 + kk*32], 128);
      bf16x16 a1 = frag_a(&Csm[(m0 + 16)*128 + kk*32], 128);
      bf16x16 b0 = frag_bt(&Bsm[(s0     )*128 + kk*32], 128);
      bf16x16 b1 = frag_bt(&Bsm[(s0 + 16)*128 + kk*32], 128);
      sacc[0][0] = wmma_bf16(a0, b0, sacc[0][0]);
      sacc[0][1] = wmma_bf16(a0, b1, sacc[0][1]);
      sacc[1][0] = wmma_bf16(a1, b0, sacc[1][0]);
      sacc[1][1] = wmma_bf16(a1, b1, sacc[1][1]);
    }
    // causal decay mask, write bf16 score tile to wave-private LDS
#pragma unroll
    for (int it = 0; it < 2; ++it)
#pragma unroll
      for (int jt = 0; jt < 2; ++jt) {
        int sloc = s0 + jt*16 + nc16;
        float as = acs[sloc];
#pragma unroll
        for (int r = 0; r < 8; ++r) {
          int l = m0 + it*16 + r + 8*hf;
          float v = sacc[it][jt][r];
          v = (l >= sloc) ? v * __expf(acs[l] - as) : 0.f;
          Ss[(it*16 + r + 8*hf)*32 + jt*16 + nc16] = (bf16)v;
        }
      }
    // Y += S * X
#pragma unroll
    for (int it = 0; it < 2; ++it) {
      bf16x16 af = frag_a(&Ss[(it*16)*32], 32);
#pragma unroll
      for (int jn = 0; jn < 4; ++jn) {
        bf16x16 bfr = frag_bt(&XsT[(jn*16)*256 + s0], 256);
        yacc[it][jn] = wmma_bf16(af, bfr, yacc[it][jn]);
      }
    }
  }
#pragma unroll
  for (int it = 0; it < 2; ++it)
#pragma unroll
    for (int jn = 0; jn < 4; ++jn)
#pragma unroll
      for (int r = 0; r < 8; ++r) {
        int l = m0 + it*16 + r + 8*hf;
        int p = jn*16 + nc16;
        ydiag[(bs0 + l)*INTER + h*HP + p] = yacc[it][jn][r];
      }

  // Phase 2: chunk state = (decay o B)^T X  -> [n=128][p=64], wave w owns n rows w*16..+15
  float alast = acs[255];
  v8f st[4];
#pragma unroll
  for (int j = 0; j < 4; ++j)
#pragma unroll
    for (int r = 0; r < 8; ++r) st[j][r] = 0.f;
  int nbase = w * 16, m = nc16;
  for (int ks = 0; ks < 8; ++ks) {
    int s0 = ks * 32;
    bf16x16 af;
#pragma unroll
    for (int e = 0; e < 16; ++e) {
      int kidx = (e < 8) ? (hf*8 + e) : (16 + hf*8 + (e - 8));
      int s = s0 + kidx;
      float v = (float)Bsm[s*128 + nbase + m] * __expf(alast - acs[s]);
      af[e] = (bf16)v;
    }
#pragma unroll
    for (int jn = 0; jn < 4; ++jn) {
      bf16x16 bfr = frag_bt(&XsT[(jn*16)*256 + s0], 256);
      st[jn] = wmma_bf16(af, bfr, st[jn]);
    }
  }
#pragma unroll
  for (int jn = 0; jn < 4; ++jn)
#pragma unroll
    for (int r = 0; r < 8; ++r) {
      int n = nbase + r + 8*hf;
      int p = jn*16 + nc16;
      cstates[((size_t)((b*NC + c)*NH + h)*HP + p)*NS + n] = st[jn][r];
    }
}

// ---------------- Inter-chunk state recurrence (16 steps) ----------------

__global__ void k_recur(const float* __restrict__ cstates, const float* __restrict__ alast_g,
                        float* __restrict__ istates) {
  int b = blockIdx.x >> 5, h = blockIdx.x & 31;
  int t = threadIdx.x;
  float prev[32];
#pragma unroll
  for (int i = 0; i < 32; ++i) prev[i] = 0.f;
  for (int z = 0; z < NC; ++z) {
    float dec = __expf(alast_g[(size_t)(b*NH + h)*NC + z]);
    size_t base = ((size_t)(b*NC + z)*NH + h) * (HP*NS);
#pragma unroll
    for (int i = 0; i < 32; ++i) {
      size_t idx = base + t + i*256;
      istates[idx] = prev[i];
      prev[i] = prev[i]*dec + cstates[idx];
    }
  }
}

// ---------------- Y_off = exp(acs) o (C * state) ; combine Y_diag + D residual ----------------

__launch_bounds__(256)
__global__ void k_yoff(const float* __restrict__ conv_out, const float* __restrict__ acs_g,
                       const float* __restrict__ istates, const float* __restrict__ Dv,
                       float* __restrict__ y) {
  extern __shared__ __align__(16) char smem[];
  float* acs = (float*)smem;            // [256]
  bf16*  Csm = (bf16*)(smem + 1024);    // [256][128]
  bf16*  St  = Csm + 256*128;           // [64][128]  state [p][n]
  int blk = blockIdx.x;
  int h = blk & 31, c = (blk >> 5) & 15, b = blk >> 9;
  int t = threadIdx.x;
  size_t bs0 = (size_t)b*SEQ + c*CHK;

  acs[t] = acs_g[((size_t)(b*NH + h)*NC + c)*CHK + t];
  for (int i = t; i < 256*128; i += 256) {
    int s = i >> 7, n = i & 127;
    Csm[i] = (bf16)conv_out[(bs0 + s)*CONVD + INTER + NS + n];
  }
  size_t sbase = ((size_t)(b*NC + c)*NH + h) * (HP*NS);
  for (int i = t; i < 64*128; i += 256) St[i] = (bf16)istates[sbase + i];
  __syncthreads();

  int w = t >> 5, lane = t & 31, nc16 = lane & 15, hf = lane >> 4;
  int m0 = w * 32;
  v8f acc[2][4];
#pragma unroll
  for (int i = 0; i < 2; ++i)
#pragma unroll
    for (int j = 0; j < 4; ++j)
#pragma unroll
      for (int r = 0; r < 8; ++r) acc[i][j][r] = 0.f;
#pragma unroll
  for (int kk = 0; kk < 4; ++kk) {
    bf16x16 a0 = frag_a(&Csm[(m0     )*128 + kk*32], 128);
    bf16x16 a1 = frag_a(&Csm[(m0 + 16)*128 + kk*32], 128);
#pragma unroll
    for (int jn = 0; jn < 4; ++jn) {
      bf16x16 bfr = frag_bt(&St[(jn*16)*128 + kk*32], 128);
      acc[0][jn] = wmma_bf16(a0, bfr, acc[0][jn]);
      acc[1][jn] = wmma_bf16(a1, bfr, acc[1][jn]);
    }
  }
  float dh = Dv[h];
#pragma unroll
  for (int it = 0; it < 2; ++it)
#pragma unroll
    for (int jn = 0; jn < 4; ++jn)
#pragma unroll
      for (int r = 0; r < 8; ++r) {
        int l = m0 + it*16 + r + 8*hf;
        int p = jn*16 + nc16;
        size_t oi = (bs0 + l)*INTER + h*HP + p;
        float xv = conv_out[(bs0 + l)*CONVD + h*HP + p];
        y[oi] = y[oi] + acc[it][jn][r]*__expf(acs[l]) + dh*xv;
      }
}

// ---------------- Gated RMSNorm (emits bf16 for the out_proj GEMM) ----------------

__global__ void k_gatenorm(const float* __restrict__ y, const float* __restrict__ proj,
                           const float* __restrict__ nw, bf16* __restrict__ yn) {
  __shared__ float red[256];
  size_t bs = blockIdx.x;
  int t = threadIdx.x;
  float vals[8], ss = 0.f;
#pragma unroll
  for (int i = 0; i < 8; ++i) {
    int col = t + i*256;
    float g = proj[bs*PROJD + col];
    float f = y[bs*INTER + col] * (g / (1.f + __expf(-g)));
    vals[i] = f;
    ss += f*f;
  }
  red[t] = ss;
  __syncthreads();
  for (int o = 128; o > 0; o >>= 1) {
    if (t < o) red[t] += red[t + o];
    __syncthreads();
  }
  float rs = rsqrtf(red[0] / (float)INTER + 1e-5f);
#pragma unroll
  for (int i = 0; i < 8; ++i) {
    int col = t + i*256;
    yn[bs*INTER + col] = (bf16)(nw[col] * vals[i] * rs);
  }
}

// ---------------- Launch ----------------

extern "C" void kernel_launch(void* const* d_in, const int* in_sizes, int n_in,
                              void* d_out, int out_size, void* d_ws, size_t ws_size,
                              hipStream_t stream) {
  (void)in_sizes; (void)n_in; (void)out_size; (void)ws_size;
  const float* x_in    = (const float*)d_in[0];
  const float* w_in    = (const float*)d_in[1];
  const float* conv_w  = (const float*)d_in[2];
  const float* conv_b  = (const float*)d_in[3];
  const float* dt_bias = (const float*)d_in[4];
  const float* A_log   = (const float*)d_in[5];
  const float* Dv      = (const float*)d_in[6];
  const float* norm_w  = (const float*)d_in[7];
  const float* w_out   = (const float*)d_in[8];
  float* out = (float*)d_out;

  char* ws = (char*)d_ws;
  size_t off = 0;
  float* proj     = (float*)(ws + off); off += (size_t)BS*PROJD*4;          // 143.7 MB
  float* conv_out = (float*)(ws + off); off += (size_t)BS*CONVD*4;          // 75.5 MB
  float* dtq      = (float*)(ws + off); off += (size_t)BS*NH*4;             // 1 MB
  float* acs      = (float*)(ws + off); off += (size_t)BB*NH*NC*CHK*4;      // 1 MB
  float* alast    = (float*)(ws + off); off += (size_t)BB*NH*NC*4;          // 4 KB
  float* y        = (float*)(ws + off); off += (size_t)BS*INTER*4;          // 67 MB
  float* cstates  = (float*)(ws + off); off += (size_t)BB*NC*NH*HP*NS*4;    // 33.6 MB
  float* istates  = (float*)(ws + off); off += (size_t)BB*NC*NH*HP*NS*4;    // 33.6 MB
  bf16* A_bf      = (bf16*)(ws + off);  off += (size_t)BS*HIDN*2;           // 16.8 MB
  bf16* WinT      = (bf16*)(ws + off);  off += (size_t)PROJP*HIDN*2;        // 9.2 MB
  bf16* WoutT     = (bf16*)(ws + off);  off += (size_t)HIDN*INTER*2;        // 4.2 MB
  bf16* yn_bf     = (bf16*)(ws + off);                                      // 33.6 MB

  // 0. one-time bf16 conversions (weights pre-transposed + zero-padded)
  k_cvt<<<(int)(((size_t)BS*HIDN + 255)/256), 256, 0, stream>>>(x_in, A_bf, (size_t)BS*HIDN);
  k_cvtT<<<(int)(((size_t)PROJP*HIDN + 255)/256), 256, 0, stream>>>(w_in, WinT, HIDN, PROJD, PROJP);
  k_cvtT<<<(int)(((size_t)HIDN*INTER + 255)/256), 256, 0, stream>>>(w_out, WoutT, INTER, HIDN, HIDN);

  // 1. in_proj GEMM: [8192,1024] x [1024,4384] (async double-buffered)
  dim3 g1(PROJP/128, BS/128);
  k_gemm_async<<<g1, 256, 0, stream>>>(A_bf, WinT, proj, BS, PROJD, HIDN);

  // 2. depthwise conv + SiLU
  k_conv<<<(int)(((size_t)BS*CONVD + 255)/256), 256, 0, stream>>>(proj, conv_w, conv_b, conv_out);

  // 3. dt softplus + per-chunk cumsum of A*dt
  k_dtscan<<<BB*NH*NC, 256, 0, stream>>>(proj, dt_bias, A_log, dtq, acs, alast);

  // 4. intra-chunk Y_diag + chunk states (WMMA)
  size_t shm_chunk = 1024 + (size_t)(64*256 + 2*256*128 + 8*32*32) * sizeof(bf16); // ~177 KB
  k_chunk<<<BB*NC*NH, 256, shm_chunk, stream>>>(conv_out, dtq, acs, y, cstates);

  // 5. inter-chunk recurrence
  k_recur<<<BB*NH, 256, 0, stream>>>(cstates, alast, istates);

  // 6. Y_off + combine with Y_diag + D residual (WMMA)
  size_t shm_yoff = 1024 + (size_t)(256*128 + 64*128) * sizeof(bf16); // ~82 KB
  k_yoff<<<BB*NC*NH, 256, shm_yoff, stream>>>(conv_out, acs, istates, Dv, y);

  // 7. gated RMSNorm -> bf16
  k_gatenorm<<<BS, 256, 0, stream>>>(y, proj, norm_w, yn_bf);

  // 8. out_proj GEMM: [8192,2048] x [2048,1024] (async double-buffered)
  dim3 g2(HIDN/128, BS/128);
  k_gemm_async<<<g2, 256, 0, stream>>>(yn_bf, WoutT, out, BS, HIDN, INTER);
}
